// DiagAttention_90778428768785
// MI455X (gfx1250) — matrix operand order repaired and checked by reading, not compiled, not run on hardware
//
#include <hip/hip_runtime.h>

typedef __attribute__((ext_vector_type(16))) _Float16 v16h;
typedef __attribute__((ext_vector_type(8)))  float    v8f;

#define WAVES 4
#define BR 64        // query rows per block (16 per wave)
#define BC 64        // key rows per iteration
#define N_TOK 4096
#define DIM 64
#define QS 72        // padded half-strides (conflict-free LDS access)
#define KS 72
#define VS 72
#define PS 72

// single-instruction pack: v_cvt_pk_rtz_f16_f32
__device__ __forceinline__ unsigned pkrtz(float a, float b) {
  typedef __fp16 fp16v2 __attribute__((ext_vector_type(2)));
  fp16v2 h = __builtin_amdgcn_cvt_pkrtz(a, b);
  return __builtin_bit_cast(unsigned, h);
}

// max across the 16 lanes of a DPP row using ROW_XMASK butterflies (pure VALU)
__device__ __forceinline__ float rowmax16(float x) {
  int xi = __builtin_bit_cast(int, x);
  int y;
  y  = __builtin_amdgcn_update_dpp(xi, xi, 0x161, 0xf, 0xf, false); // XMASK 1
  x  = fmaxf(x, __builtin_bit_cast(float, y)); xi = __builtin_bit_cast(int, x);
  y  = __builtin_amdgcn_update_dpp(xi, xi, 0x162, 0xf, 0xf, false); // XMASK 2
  x  = fmaxf(x, __builtin_bit_cast(float, y)); xi = __builtin_bit_cast(int, x);
  y  = __builtin_amdgcn_update_dpp(xi, xi, 0x164, 0xf, 0xf, false); // XMASK 4
  x  = fmaxf(x, __builtin_bit_cast(float, y)); xi = __builtin_bit_cast(int, x);
  y  = __builtin_amdgcn_update_dpp(xi, xi, 0x168, 0xf, 0xf, false); // XMASK 8
  x  = fmaxf(x, __builtin_bit_cast(float, y));
  return x;
}

__global__ __launch_bounds__(128)
void diag_flash_attn(const float* __restrict__ x, const float* __restrict__ qd,
                     const float* __restrict__ kd, const float* __restrict__ vd,
                     float* __restrict__ out) {
  __shared__ _Float16 lds_q[BR * QS];
  __shared__ _Float16 lds_k[BC * KS];
  __shared__ _Float16 lds_v[DIM * VS];          // column-major: [d][key]
  __shared__ _Float16 lds_p[WAVES][16 * PS];

  const int t    = threadIdx.x;
  const int lane = t & 31;
  const int w    = t >> 5;
  const int col  = lane & 15;
  const int hi   = lane >> 4;

  const int nb = N_TOK / BR;                  // 64
  const int b  = blockIdx.x / nb;
  const int q0 = (blockIdx.x % nb) * BR;
  const float* xb = x + (size_t)b * N_TOK * DIM;

  const float SCL = 0.125f * 1.44269504088896340736f; // 1/sqrt(64) * log2(e)

  // ---- Q block -> LDS (f16, scaled by q_diag * SCL) --------------------
  {
    int rq  = t >> 1;                 // 0..63
    int cq0 = (t & 1) * 32;
    const float4* src  = (const float4*)(xb + (size_t)(q0 + rq) * DIM + cq0);
    const float4* qdv4 = (const float4*)(qd + cq0);
    unsigned* dst = (unsigned*)&lds_q[rq * QS + cq0];
#pragma unroll
    for (int u = 0; u < 8; ++u) {
      float4 xv = src[u];
      float4 dv = qdv4[u];
      dst[u*2+0] = pkrtz(xv.x*dv.x*SCL, xv.y*dv.y*SCL);
      dst[u*2+1] = pkrtz(xv.z*dv.z*SCL, xv.w*dv.w*SCL);
    }
  }

  // ---- K/V fill assignment: key-row pairs {rv,rv+1,rv+32,rv+33} x 8 cols
  const int rv  = ((t >> 3) & 15) * 2;   // 0,2,...,30
  const int c0v = (t & 7) * 8;           // 0,8,...,56
  float kdr[8], vdr[8];
#pragma unroll
  for (int u = 0; u < 2; ++u) {
    float4 kv = ((const float4*)(kd + c0v))[u];
    float4 vv = ((const float4*)(vd + c0v))[u];
    kdr[u*4+0]=kv.x; kdr[u*4+1]=kv.y; kdr[u*4+2]=kv.z; kdr[u*4+3]=kv.w;
    vdr[u*4+0]=vv.x; vdr[u*4+1]=vv.y; vdr[u*4+2]=vv.z; vdr[u*4+3]=vv.w;
  }

  __syncthreads();

  // ---- Q A-fragments (16x32 f16 A layout), two d-chunks ----------------
  v16h qf[2];
#pragma unroll
  for (int f = 0; f < 2; ++f) {
    union { v16h v; unsigned u[8]; } tmp;
#pragma unroll
    for (int v = 0; v < 8; ++v) {
      int k = (v < 4) ? (2*v + 8*hi) : (16 + 2*(v-4) + 8*hi);
      tmp.u[v] = *(const unsigned*)&lds_q[(w*16 + col) * QS + f*32 + k];
    }
    qf[f] = tmp.v;
  }

  // all-ones B operand (32x16) for WMMA row sums
  v16h ones;
#pragma unroll
  for (int i = 0; i < 16; ++i) ones[i] = (_Float16)1.0f;

  v8f o0 = {}, o1 = {}, o2 = {}, o3 = {}, lacc = {};
  float m_i[8];
#pragma unroll
  for (int i = 0; i < 8; ++i) m_i[i] = -3.0e38f;

  // LDS fragment loaders -------------------------------------------------
  auto ldK = [&](int rowbase, int dc) -> v16h {          // B frags from row-major K
    union { v16h v; unsigned u[8]; } tp;
#pragma unroll
    for (int v = 0; v < 8; ++v) {
      int d = dc*32 + ((v < 4) ? (2*v + 8*hi) : (16 + 2*(v-4) + 8*hi));
      tp.u[v] = *(const unsigned*)&lds_k[(rowbase + col) * KS + d];
    }
    return tp.v;
  };
  auto ldV = [&](int dt, int kc) -> v16h {               // B frags from col-major V
    union { v16h v; unsigned u[8]; } tp;
#pragma unroll
    for (int v = 0; v < 8; ++v) {
      int k = kc*32 + ((v < 4) ? (2*v + 8*hi) : (16 + 2*(v-4) + 8*hi));
      tp.u[v] = *(const unsigned*)&lds_v[(dt*16 + col) * VS + k];
    }
    return tp.v;
  };
  auto ldP = [&](int kc) -> v16h {                       // A frag from per-wave P tile
    union { v16h v; unsigned u[8]; } tp;
#pragma unroll
    for (int v = 0; v < 8; ++v) {
      int k = kc*32 + ((v < 4) ? (2*v + 8*hi) : (16 + 2*(v-4) + 8*hi));
      tp.u[v] = *(const unsigned*)&lds_p[w][col * PS + k];
    }
    return tp.v;
  };

  // register prefetch of the shared x block (K and V both derive from it)
  float xr0[8], xr1[8], xr2[8], xr3[8];
  auto ld8 = [&](const float* p, float* d) {
#pragma unroll
    for (int u = 0; u < 2; ++u) {
      float4 a = ((const float4*)p)[u];
      d[u*4+0]=a.x; d[u*4+1]=a.y; d[u*4+2]=a.z; d[u*4+3]=a.w;
    }
  };
  auto prefetch = [&](int base) {
    ld8(xb + (size_t)(base + rv)      * DIM + c0v, xr0);
    ld8(xb + (size_t)(base + rv + 1)  * DIM + c0v, xr1);
    ld8(xb + (size_t)(base + rv + 32) * DIM + c0v, xr2);
    ld8(xb + (size_t)(base + rv + 33) * DIM + c0v, xr3);
  };
  prefetch(0);

  const int T = N_TOK / BC;  // 64 key blocks
  for (int j = 0; j < T; ++j) {
    __syncthreads();
    // ---- fill LDS: K row-major, V col-major, both packed b32 stores ----
    {
      auto kfill = [&](int row, const float* xv) {
        unsigned* kdst = (unsigned*)&lds_k[row * KS + c0v];
#pragma unroll
        for (int u = 0; u < 4; ++u)
          kdst[u] = pkrtz(xv[2*u]*kdr[2*u], xv[2*u+1]*kdr[2*u+1]);
      };
      kfill(rv,      xr0);
      kfill(rv + 1,  xr1);
      kfill(rv + 32, xr2);
      kfill(rv + 33, xr3);
#pragma unroll
      for (int d = 0; d < 8; ++d) {
        int c = c0v + d;
        *(unsigned*)&lds_v[c * VS + rv]      = pkrtz(xr0[d]*vdr[d], xr1[d]*vdr[d]);
        *(unsigned*)&lds_v[c * VS + rv + 32] = pkrtz(xr2[d]*vdr[d], xr3[d]*vdr[d]);
      }
    }
    __syncthreads();
    // prefetch next block (overlaps compute)
    if (j + 1 < T) prefetch((j + 1) * BC);

    // ---- S = Q K^T : four 16x16 f32 accumulator tiles ------------------
    v8f s0 = {}, s1 = {}, s2 = {}, s3 = {};
#pragma unroll
    for (int c = 0; c < 2; ++c) {
      v16h b0 = ldK( 0, c), b1 = ldK(16, c), b2 = ldK(32, c), b3 = ldK(48, c);
      s0 = __builtin_amdgcn_wmma_f32_16x16x32_f16(false, qf[c], false, b0, (short)0, s0, false, false);
      s1 = __builtin_amdgcn_wmma_f32_16x16x32_f16(false, qf[c], false, b1, (short)0, s1, false, false);
      s2 = __builtin_amdgcn_wmma_f32_16x16x32_f16(false, qf[c], false, b2, (short)0, s2, false, false);
      s3 = __builtin_amdgcn_wmma_f32_16x16x32_f16(false, qf[c], false, b3, (short)0, s3, false, false);
    }

    // ---- online softmax: DPP row-max, exp2, rescale accumulators -------
#pragma unroll
    for (int i = 0; i < 8; ++i) {
      float a = fmaxf(fmaxf(s0[i], s1[i]), fmaxf(s2[i], s3[i]));
      a = rowmax16(a);
      float mnew = fmaxf(m_i[i], a);
      float corr = __builtin_amdgcn_exp2f(m_i[i] - mnew);
      m_i[i] = mnew;
      float p0 = __builtin_amdgcn_exp2f(s0[i] - mnew);
      float p1 = __builtin_amdgcn_exp2f(s1[i] - mnew);
      float p2 = __builtin_amdgcn_exp2f(s2[i] - mnew);
      float p3 = __builtin_amdgcn_exp2f(s3[i] - mnew);
      lacc[i] *= corr;
      o0[i] *= corr; o1[i] *= corr; o2[i] *= corr; o3[i] *= corr;
      int mrow = i + 8*hi;
      _Float16* pr = &lds_p[w][mrow * PS + col];
      pr[0]  = (_Float16)p0;
      pr[16] = (_Float16)p1;
      pr[32] = (_Float16)p2;
      pr[48] = (_Float16)p3;
    }

    // ---- P fragments; row sums and O += P*V all via WMMA ---------------
    v16h pa0 = ldP(0), pa1 = ldP(1);
    lacc = __builtin_amdgcn_wmma_f32_16x16x32_f16(false, pa0, false, ones, (short)0, lacc, false, false);
    lacc = __builtin_amdgcn_wmma_f32_16x16x32_f16(false, pa1, false, ones, (short)0, lacc, false, false);
#pragma unroll
    for (int dt = 0; dt < 4; ++dt) {
      v16h bv0 = ldV(dt, 0), bv1 = ldV(dt, 1);
      v8f& o = dt==0 ? o0 : dt==1 ? o1 : dt==2 ? o2 : o3;
      o = __builtin_amdgcn_wmma_f32_16x16x32_f16(false, pa0, false, bv0, (short)0, o, false, false);
      o = __builtin_amdgcn_wmma_f32_16x16x32_f16(false, pa1, false, bv1, (short)0, o, false, false);
    }
  }

  // ---- epilogue: normalize by WMMA-accumulated row sums, store f32 -----
  float* ob = out + (size_t)b * N_TOK * DIM;
#pragma unroll
  for (int i = 0; i < 8; ++i) {
    float inv = 1.0f / lacc[i];
    int row = q0 + w*16 + i + 8*hi;
    float* orow = ob + (size_t)row * DIM + col;
    orow[0]  = o0[i] * inv;
    orow[16] = o1[i] * inv;
    orow[32] = o2[i] * inv;
    orow[48] = o3[i] * inv;
  }
}

extern "C" void kernel_launch(void* const* d_in, const int* in_sizes, int n_in,
                              void* d_out, int out_size, void* d_ws, size_t ws_size,
                              hipStream_t stream) {
  const float* x  = (const float*)d_in[0];
  const float* qd = (const float*)d_in[1];
  const float* kd = (const float*)d_in[2];
  const float* vd = (const float*)d_in[3];
  float* out = (float*)d_out;
  const int B = 4;
  dim3 grid(B * (N_TOK / BR));   // 256 blocks
  dim3 block(128);               // 4 waves of 32
  diag_flash_attn<<<grid, block, 0, stream>>>(x, qd, kd, vd, out);
}
